// GCN_39883066310757
// MI455X (gfx1250) — matrix operand order repaired and checked by reading, not compiled, not run on hardware
//
#include <hip/hip_runtime.h>

// GIN forward on MI455X (gfx1250): bf16 WMMA GEMMs, f32 master activations.
// Memory-bound workload (~6 GB traffic @ 23.3 TB/s); GEMMs pushed onto
// v_wmma_f32_16x16x32_bf16 so matmul compute is off the critical path.
// v2: B fragments batched per k-step (b[4]) so the compiler can stagger
// s_wait_loadcnt and overlap global B loads with WMMA issue.

#define NNODES 50000
#define NEDGES 150000
#define DIM    512
#define NLAYER 4
#define NND    ((size_t)NNODES * DIM)        // 25,600,000
#define WELEM  ((size_t)NLAYER * DIM * DIM)  // 4*512*512 = 1,048,576
#define BN_EPS 1e-5f

typedef __attribute__((ext_vector_type(16))) __bf16 v16bf;
typedef __attribute__((ext_vector_type(8)))  float  v8f;

union BFrag { v16bf v; uint4 q[2]; };

// ---- f32 -> bf16 round-to-nearest-even ----
__device__ __forceinline__ unsigned short f2bf(float f) {
  unsigned u = __builtin_bit_cast(unsigned, f);
  unsigned r = u + 0x7FFFu + ((u >> 16) & 1u);
  return (unsigned short)(r >> 16);
}

// ---- zero n floats ----
__global__ void k_zero_f32(float* __restrict__ p, int n) {
  int i = blockIdx.x * blockDim.x + threadIdx.x;
  if (i < n) p[i] = 0.0f;
}

// ---- out[i] *= s ----
__global__ void k_scale_f32(float* __restrict__ p, int n, float s) {
  int i = blockIdx.x * blockDim.x + threadIdx.x;
  if (i < n) p[i] *= s;
}

// ---- float4 copy (sum <- h) ----
__global__ void k_copy4(const float4* __restrict__ in, float4* __restrict__ out, size_t n4) {
  size_t i = (size_t)blockIdx.x * blockDim.x + threadIdx.x;
  if (i < n4) out[i] = in[i];
}

// ---- edge scatter: sum[dst] += h[src], one block (128 thr) per edge ----
__global__ void __launch_bounds__(128) k_edge_scatter(
    const float* __restrict__ h, const int* __restrict__ src,
    const int* __restrict__ dst, float* __restrict__ sum) {
  int e = blockIdx.x;
  int s = src[e], d = dst[e];
  float4 v = ((const float4*)(h + (size_t)s * DIM))[threadIdx.x];
  float* o = sum + (size_t)d * DIM + (size_t)threadIdx.x * 4;
  atomicAdd(o + 0, v.x); atomicAdd(o + 1, v.y);
  atomicAdd(o + 2, v.z); atomicAdd(o + 3, v.w);
}

// ---- f32 -> bf16 (4 elems/thread) ----
__global__ void k_cvt_bf16(const float* __restrict__ in, unsigned short* __restrict__ out, size_t n) {
  size_t i = ((size_t)blockIdx.x * blockDim.x + threadIdx.x) * 4;
  if (i >= n) return;
  float4 v = *(const float4*)(in + i);
  unsigned p0 = (unsigned)f2bf(v.x) | ((unsigned)f2bf(v.y) << 16);
  unsigned p1 = (unsigned)f2bf(v.z) | ((unsigned)f2bf(v.w) << 16);
  uint2 p; p.x = p0; p.y = p1;
  *(uint2*)(out + i) = p;
}

// ---- core: Out[m,n] = sum_k A_bf16[m,k] * W_bf16[n,k] + bias[n]  (z = s @ W^T + b)
// One block per 16-row M tile; 8 waves cover N=512 (64 cols each, 4 acc tiles);
// A tile staged in LDS (padded stride 520 halfwords to spread banks).
__global__ void __launch_bounds__(256) k_gemm_bf16(
    const unsigned short* __restrict__ A,   // [M,512] bf16
    const unsigned short* __restrict__ W,   // [512,512] bf16, row n holds K contiguous
    const float* __restrict__ bias,         // [512]
    float* __restrict__ Out) {              // [M,512] f32
  __shared__ unsigned short sA[16 * 520];   // 16.25 KB
  const int m0  = blockIdx.x * 16;
  const int tid = threadIdx.x;

  // cooperative A-tile load: 16 rows x 512 = 1024 chunks of 8 halfwords
#pragma unroll
  for (int i = 0; i < 4; ++i) {
    int c    = tid + i * 256;
    int row  = c >> 6;
    int col8 = (c & 63) << 3;
    uint4 v = *(const uint4*)(A + (size_t)(m0 + row) * DIM + col8);
    *(uint4*)(&sA[row * 520 + col8]) = v;
  }
  __syncthreads();

  const int wave = tid >> 5;
  const int lane = tid & 31;
  const int lh   = lane & 15;   // N within tile / M row for A frag
  const int hi   = lane >> 4;   // half-wave select
  const int n0   = wave * 64;

  v8f acc[4];
#pragma unroll
  for (int t = 0; t < 4; ++t) {
    float bv = bias[n0 + t * 16 + lh];
#pragma unroll
    for (int r = 0; r < 8; ++r) acc[t][r] = bv;
  }

  // loop-invariant base pointer for this lane's B rows
  const unsigned short* wbase = W + (size_t)(n0 + lh) * DIM + hi * 8;

#pragma unroll 4
  for (int ks = 0; ks < 16; ++ks) {
    // A frag (16x32): lane lh = row M; chunks K[ko..ko+7], K[ko+16..ko+23]
    BFrag a;
    int ko = ks * 32 + hi * 8;
    a.q[0] = *(const uint4*)(&sA[lh * 520 + ko]);
    a.q[1] = *(const uint4*)(&sA[lh * 520 + ko + 16]);

    // B frags (32x16 each): issue all 8 b128 loads first so the compiler can
    // stagger s_wait_loadcnt and keep WMMAs flowing.
    BFrag b[4];
#pragma unroll
    for (int t = 0; t < 4; ++t) {
      const unsigned short* wp = wbase + (size_t)t * 16 * DIM + ks * 32;
      b[t].q[0] = *(const uint4*)(wp);
      b[t].q[1] = *(const uint4*)(wp + 16);
    }
#pragma unroll
    for (int t = 0; t < 4; ++t) {
      acc[t] = __builtin_amdgcn_wmma_f32_16x16x32_bf16(
          false, a.v, false, b[t].v, (short)0, acc[t], false, false);
    }
  }

  // C/D layout: VGPR r, lanes 0-15 -> M=r; lanes 16-31 -> M=r+8; N = lane&15
#pragma unroll
  for (int t = 0; t < 4; ++t) {
    int n = n0 + t * 16 + lh;
#pragma unroll
    for (int r = 0; r < 8; ++r) {
      int m = m0 + r + hi * 8;
      Out[(size_t)m * DIM + n] = acc[t][r];
    }
  }
}

// ---- column sum + sum of squares (512 threads = 1 col each) ----
__global__ void __launch_bounds__(512) k_colstats(
    const float* __restrict__ z, float* __restrict__ cs, float* __restrict__ csq,
    int rows, int rpb) {
  int n  = threadIdx.x;
  int r0 = blockIdx.x * rpb;
  int r1 = r0 + rpb; if (r1 > rows) r1 = rows;
  float s = 0.f, q = 0.f;
  for (int r = r0; r < r1; ++r) {
    float v = z[(size_t)r * DIM + n];
    s += v; q = fmaf(v, v, q);
  }
  atomicAdd(&cs[n], s);
  atomicAdd(&csq[n], q);
}

// ---- fold BN into scale/shift: sc = gamma*rsqrt(var+eps), sh = beta - mu*sc ----
__global__ void __launch_bounds__(512) k_bnprep(
    const float* __restrict__ cs, const float* __restrict__ csq,
    const float* __restrict__ gamma, const float* __restrict__ beta,
    float* __restrict__ scale, float* __restrict__ shift) {
  int n = threadIdx.x;
  const float inv = 1.0f / (float)NNODES;
  float mu  = cs[n] * inv;
  float var = csq[n] * inv - mu * mu;   // biased variance (matches reference)
  float sc  = gamma[n] * rsqrtf(var + BN_EPS);
  scale[n] = sc;
  shift[n] = beta[n] - mu * sc;
}

// ---- fused BN + ReLU + bf16 convert (4 elems/thread) ----
__global__ void k_bn_relu_cvt(
    const float* __restrict__ z, const float* __restrict__ scale,
    const float* __restrict__ shift, unsigned short* __restrict__ out, size_t n) {
  size_t i = ((size_t)blockIdx.x * blockDim.x + threadIdx.x) * 4;
  if (i >= n) return;
  int c = (int)(i & (DIM - 1));
  float4 v  = *(const float4*)(z + i);
  float4 sc = *(const float4*)(scale + c);
  float4 sh = *(const float4*)(shift + c);
  float r0 = fmaxf(fmaf(v.x, sc.x, sh.x), 0.f);
  float r1 = fmaxf(fmaf(v.y, sc.y, sh.y), 0.f);
  float r2 = fmaxf(fmaf(v.z, sc.z, sh.z), 0.f);
  float r3 = fmaxf(fmaf(v.w, sc.w, sh.w), 0.f);
  unsigned p0 = (unsigned)f2bf(r0) | ((unsigned)f2bf(r1) << 16);
  unsigned p1 = (unsigned)f2bf(r2) | ((unsigned)f2bf(r3) << 16);
  uint2 p; p.x = p0; p.y = p1;
  *(uint2*)(out + i) = p;
}

// ---- accumulate column sums into out[0..511] (for final means) ----
__global__ void __launch_bounds__(512) k_colaccum(
    const float* __restrict__ h, float* __restrict__ out, int rows, int rpb) {
  int n  = threadIdx.x;
  int r0 = blockIdx.x * rpb;
  int r1 = r0 + rpb; if (r1 > rows) r1 = rows;
  float s = 0.f;
  for (int r = r0; r < r1; ++r) s += h[(size_t)r * DIM + n];
  atomicAdd(&out[n], s);
}

extern "C" void kernel_launch(void* const* d_in, const int* in_sizes, int n_in,
                              void* d_out, int out_size, void* d_ws, size_t ws_size,
                              hipStream_t stream) {
  const float* x     = (const float*)d_in[0];
  const float* W1    = (const float*)d_in[1];
  const float* b1    = (const float*)d_in[2];
  const float* gamma = (const float*)d_in[3];
  const float* beta  = (const float*)d_in[4];
  const float* W2    = (const float*)d_in[5];
  const float* b2    = (const float*)d_in[6];
  const int*   src   = (const int*)d_in[7];
  const int*   dst   = (const int*)d_in[8];
  float* out = (float*)d_out;

  // workspace layout (all offsets 16B aligned): ~260 MB
  float* buf_h = (float*)d_ws;                       // [N,D] f32 current h / GEMM2 out
  float* buf_s = buf_h + NND;                        // [N,D] f32 sum, then z
  unsigned short* buf_hb = (unsigned short*)(buf_s + NND);  // [N,D] bf16 operand
  unsigned short* w1b = buf_hb + NND;                // [4,512,512] bf16
  unsigned short* w2b = w1b + WELEM;                 // [4,512,512] bf16
  float* stats   = (float*)(w2b + WELEM);
  float* colsum  = stats;
  float* colsumsq = stats + 512;
  float* scale   = stats + 1024;
  float* shift   = stats + 1536;

  const size_t nd4 = NND / 4;                        // 6,400,000
  dim3 cg((unsigned)((nd4 + 255) / 256));
  dim3 wg((unsigned)((WELEM / 4 + 255) / 256));

  // one-time: weights to bf16; zero output accumulators
  k_cvt_bf16<<<wg, 256, 0, stream>>>(W1, w1b, WELEM);
  k_cvt_bf16<<<wg, 256, 0, stream>>>(W2, w2b, WELEM);
  k_zero_f32<<<4, 256, 0, stream>>>(out, 1024);

  const float* h = x;
  for (int L = 0; L < NLAYER; ++L) {
    // s = h + segment_sum(h[src] -> dst)
    k_copy4<<<cg, 256, 0, stream>>>((const float4*)h, (float4*)buf_s, nd4);
    k_edge_scatter<<<NEDGES, 128, 0, stream>>>(h, src, dst, buf_s);
    // hb = bf16(s)
    k_cvt_bf16<<<cg, 256, 0, stream>>>(buf_s, buf_hb, NND);
    // z = hb @ W1^T + b1  (overwrites buf_s; hb already captured)
    k_gemm_bf16<<<NNODES / 16, 256, 0, stream>>>(buf_hb, w1b + (size_t)L * DIM * DIM,
                                                 b1 + L * DIM, buf_s);
    // batch stats -> scale/shift
    k_zero_f32<<<4, 256, 0, stream>>>(stats, 1024);
    k_colstats<<<200, 512, 0, stream>>>(buf_s, colsum, colsumsq, NNODES, 250);
    k_bnprep<<<1, 512, 0, stream>>>(colsum, colsumsq, gamma + L * DIM, beta + L * DIM,
                                    scale, shift);
    // zb = bf16(relu(z*sc + sh))
    k_bn_relu_cvt<<<cg, 256, 0, stream>>>(buf_s, scale, shift, buf_hb, NND);
    // h_next = zb @ W2^T + b2
    k_gemm_bf16<<<NNODES / 16, 256, 0, stream>>>(buf_hb, w2b + (size_t)L * DIM * DIM,
                                                 b2 + L * DIM, buf_h);
    // graph readout accumulation: out[0:512]=mean(layer 3), out[512:1024]=mean(layer 2)
    if (L == 2) k_colaccum<<<200, 512, 0, stream>>>(buf_h, out + 512, NNODES, 250);
    if (L == 3) k_colaccum<<<200, 512, 0, stream>>>(buf_h, out, NNODES, 250);
    h = buf_h;
  }
  k_scale_f32<<<4, 256, 0, stream>>>(out, 1024, 1.0f / (float)NNODES);
}
